// FlatCachedAdapterEmbedding_12721693130776
// MI455X (gfx1250) — compile-verified, matrix operand order checked by reading, not run on hardware
//
#include <hip/hip_runtime.h>
#include <stdint.h>

typedef __attribute__((ext_vector_type(16))) _Float16 v16h;
typedef __attribute__((ext_vector_type(8)))  float    v8f;
typedef __attribute__((ext_vector_type(4)))  float    v4f;

#define D_MODEL 4096
#define RANK    32
#define LORA_SCALING 0.5f   // alpha / rank = 16 / 32, exact in fp
#define M_ITER  8           // M-tiles per wave
#define N_TILES (D_MODEL / 16)        // 256
#define WAVES_PER_BLOCK 8

// out[t, d] = (float)w_base_q[ids[t]*D + d] * w_scale
//           + sum_k A[ids[t], k] * (B[k, d] * 0.5)
// One V_WMMA_F32_16X16X32_F16 per 16(token) x 16(dim) tile, K = 32 = RANK.
__global__ __launch_bounds__(256)
void lora_embed_wmma_kernel(const int*    __restrict__ ids,
                            const int8_t* __restrict__ wq,
                            const float*  __restrict__ wscale,
                            const float*  __restrict__ A,
                            const float*  __restrict__ B,
                            float*        __restrict__ out)
{
    const int lane    = threadIdx.x & 31;
    const int waveBlk = threadIdx.x >> 5;
    const int waveId  = blockIdx.x * WAVES_PER_BLOCK + waveBlk;
    const int n_tile  = waveId & (N_TILES - 1);    // 0..255
    // wave-uniform by construction; make it provably uniform for the backend
    const int m_chunk = __builtin_amdgcn_readfirstlane(waveId >> 8);  // 0..127
    const int n0      = n_tile * 16;
    const int col     = n0 + (lane & 15);          // this lane's N (dim) index
    // 16-bit A/B fragment K layout (wave32): lanes 0-15 hold K {0..7, 16..23},
    // lanes 16-31 hold K {8..15, 24..31}; two halves packed per VGPR.
    const int  k_base  = (lane & 16) ? 8 : 0;
    const bool hi_half = (lane & 16) != 0;         // C/D rows +8 for lanes 16-31
    const float scale  = wscale[0];

    // ---- B fragment: column N=col of B (32 x 4096), scaled by alpha/rank.
    // Loaded once per wave; B is 512 KB and L2-resident.
    v16h bfrag;
#pragma unroll
    for (int j = 0; j < 8; ++j) {
        bfrag[j]     = (_Float16)(B[(k_base + j)      * D_MODEL + col] * LORA_SCALING);
        bfrag[j + 8] = (_Float16)(B[(k_base + 16 + j) * D_MODEL + col] * LORA_SCALING);
    }

#pragma unroll 1
    for (int it = 0; it < M_ITER; ++it) {
        const int t0   = (m_chunk * M_ITER + it) * 16;
        const int* idp = ids + t0;                 // wave-uniform address

        // Lane L holds the token id for A-fragment row M = L & 15.
        const int id_self = idp[lane & 15];

        // ---- A fragment: row id_self of adapter_A (contiguous 32 floats).
        const float* arow = A + (size_t)id_self * RANK + k_base;
        const v4f a0 = *(const v4f*)(arow);        // K k_base+0 .. +3
        const v4f a1 = *(const v4f*)(arow + 4);    // K k_base+4 .. +7
        const v4f a2 = *(const v4f*)(arow + 16);   // K k_base+16 .. +19
        const v4f a3 = *(const v4f*)(arow + 20);   // K k_base+20 .. +23
        v16h afrag;
#pragma unroll
        for (int j = 0; j < 4; ++j) {
            afrag[j]      = (_Float16)a0[j];
            afrag[j + 4]  = (_Float16)a1[j];
            afrag[j + 8]  = (_Float16)a2[j];
            afrag[j + 12] = (_Float16)a3[j];
        }

        // ---- LoRA tile: D = A(16x32) x B(32x16) + 0, f32 accumulate.
        v8f acc = {};
        acc = __builtin_amdgcn_wmma_f32_16x16x32_f16(
            /*neg_a=*/false, afrag, /*neg_b=*/false, bfrag,
            /*c_mod=*/(short)0, acc, /*reuse_a=*/false, /*reuse_b=*/false);

        // ---- Fuse dequantized int8 base gather and store.
        // C/D layout: VGPR v -> (M = v,   N = lane&15) for lanes 0-15,
        //             (M = v+8, N = lane&15) for lanes 16-31.
        float* orow = out + (size_t)(t0 + (hi_half ? 8 : 0)) * D_MODEL + col;
#pragma unroll
        for (int v = 0; v < 8; ++v) {
            // ids at uniform addresses; per-half row id picked by one cndmask
            const int id_lo = idp[v];
            const int id_hi = idp[v + 8];
            const int idm   = hi_half ? id_hi : id_lo;
            // 32-bit gather offset (max ~2.06e8 < 2^31) -> saddr+voffset form
            const uint32_t off = (uint32_t)idm * (uint32_t)D_MODEL + (uint32_t)col;
            const float base = (float)wq[off] * scale;
            orow[(size_t)v * D_MODEL] = acc[v] + base;
        }
    }
}

extern "C" void kernel_launch(void* const* d_in, const int* in_sizes, int n_in,
                              void* d_out, int out_size, void* d_ws, size_t ws_size,
                              hipStream_t stream) {
    (void)in_sizes; (void)n_in; (void)out_size; (void)d_ws; (void)ws_size;
    const int*    ids    = (const int*)   d_in[0];  // [4, 4096] int32
    const int8_t* wq     = (const int8_t*)d_in[1];  // [50257, 4096] int8
    const float*  wscale = (const float*) d_in[2];  // [1] f32
    const float*  A      = (const float*) d_in[3];  // [50257, 32] f32
    const float*  B      = (const float*) d_in[4];  // [32, 4096] f32
    float*        out    = (float*)       d_out;    // [4, 4096, 4096] f32

    // 16384 tokens / 16 = 1024 M-tiles; 8 per wave -> 128 M-chunks.
    // 256 N-tiles * 128 M-chunks = 32768 waves / 8 per block = 4096 blocks.
    const int blocks = (N_TILES * (1024 / M_ITER)) / WAVES_PER_BLOCK;
    lora_embed_wmma_kernel<<<blocks, 32 * WAVES_PER_BLOCK, 0, stream>>>(
        ids, wq, wscale, A, B, out);
}
